// BaseRNN_60206851555806
// MI455X (gfx1250) — matrix operand order repaired and checked by reading, not compiled, not run on hardware
//
#include <hip/hip_runtime.h>
#include <hip/hip_bf16.h>

// Problem constants (from reference): B=64, S=1024, I=1024, H=1024
#define B_    64
#define S_    1024
#define I_    1024
#define H_    1024
#define NBLK  64      // H_/16 column groups, one per workgroup (one WGP each)
#define TPB   128     // 4 waves of 32 (wave32): wave w owns batch rows 16w..16w+15

typedef __attribute__((ext_vector_type(16))) __bf16 v16bf;
typedef __attribute__((ext_vector_type(8)))  float  v8f;

union Frag16 { v16bf v; uint4 q[2]; };   // 32 bytes: one WMMA 16-bit A or B operand

// fp32 -> bf16, round-to-nearest-even (pure integer; no __bf16 arithmetic needed)
__device__ __forceinline__ unsigned short f2bf(float f) {
  unsigned u = __float_as_uint(f);
  u += 0x7fffu + ((u >> 16) & 1u);
  return (unsigned short)(u >> 16);
}

__device__ __forceinline__ uint4 pack8bf(float4 a, float4 b) {
  union { unsigned short h[8]; uint4 q; } u;
  u.h[0] = f2bf(a.x); u.h[1] = f2bf(a.y); u.h[2] = f2bf(a.z); u.h[3] = f2bf(a.w);
  u.h[4] = f2bf(b.x); u.h[5] = f2bf(b.y); u.h[6] = f2bf(b.z); u.h[7] = f2bf(b.w);
  return u.q;
}

__device__ __forceinline__ float sig_(float x)  { return 1.0f / (1.0f + __expf(-x)); }
__device__ __forceinline__ float tanh_(float x) { return 1.0f - 2.0f / (__expf(2.0f * x) + 1.0f); }

// ---- One-shot streaming pre-pass: x fp32 -> bf16 (done once, not 64x per block per step)
extern "C" __global__ __launch_bounds__(256)
void x_to_bf16(const float* __restrict__ x, unsigned short* __restrict__ xb) {
  const size_t n = (size_t)B_ * S_ * I_;
  const size_t stride = (size_t)gridDim.x * blockDim.x * 8;
  for (size_t i = ((size_t)blockIdx.x * blockDim.x + threadIdx.x) * 8; i < n; i += stride) {
    float4 a = *(const float4*)(x + i);
    float4 b = *(const float4*)(x + i + 4);
    *(uint4*)(xb + i) = pack8bf(a, b);     // 8 bf16 = 16B store
  }
}

extern "C" __global__ __launch_bounds__(TPB, 1)
void lstm_persistent(const float* __restrict__ x,      // (B,S,I) fp32
                     const float* __restrict__ Wih,    // (4H,I)
                     const float* __restrict__ Whh,    // (4H,H)
                     const float* __restrict__ bih,    // (4H)
                     const float* __restrict__ bhh,    // (4H)
                     float* __restrict__ out,          // (B,S,H)
                     unsigned short* __restrict__ h0buf,  // (B,H) bf16 ping
                     unsigned short* __restrict__ h1buf,  // (B,H) bf16 pong
                     const unsigned short* __restrict__ xbf,  // (B,S,I) bf16 or nullptr
                     unsigned* __restrict__ barCnt,
                     unsigned* __restrict__ barGen)
{
  // Dynamic LDS: Whh slice (64x1024 bf16 = 128KB) + Wih slice (128KB) + bias sums
  extern __shared__ unsigned char smem[];
  unsigned short* Whh_s = (unsigned short*)smem;
  unsigned short* Wih_s = Whh_s + 64 * 1024;
  float*          bsum  = (float*)(Wih_s + 64 * 1024);

  const int tid  = threadIdx.x;
  const int bc   = blockIdx.x;        // which group of 16 h-columns we own
  const int n0   = bc * 16;
  const int wave = tid >> 5;
  const int lane = tid & 31;
  const int half = lane >> 4;         // lane group (0: lanes 0-15, 1: lanes 16-31)
  const int lm   = lane & 15;
  const int m0   = wave * 16;         // batch-row tile base for this wave

  // ---- Stage this block's weight slices into LDS (once; resident for all 1024 steps)
  // Gate row g*16+j in LDS  <->  global gate row g*H + n0 + j   (g = 0..3 : i,f,g,o)
  for (int idx = tid * 4; idx < 64 * 1024; idx += TPB * 4) {
    const int r = idx >> 10, k = idx & 1023;
    const int grow = (r >> 4) * H_ + n0 + (r & 15);
    {
      float4 v = *(const float4*)(Whh + (size_t)grow * H_ + k);
      union { unsigned short h[4]; uint2 q; } u;
      u.h[0] = f2bf(v.x); u.h[1] = f2bf(v.y); u.h[2] = f2bf(v.z); u.h[3] = f2bf(v.w);
      *(uint2*)(Whh_s + idx) = u.q;
    }
    {
      float4 v = *(const float4*)(Wih + (size_t)grow * I_ + k);
      union { unsigned short h[4]; uint2 q; } u;
      u.h[0] = f2bf(v.x); u.h[1] = f2bf(v.y); u.h[2] = f2bf(v.z); u.h[3] = f2bf(v.w);
      *(uint2*)(Wih_s + idx) = u.q;
    }
  }
  if (tid < 64) {
    const int grow = (tid >> 4) * H_ + n0 + (tid & 15);
    bsum[tid] = bih[grow] + bhh[grow];
  }
  __syncthreads();

  const int row = m0 + lm;                               // batch row this lane loads (A operand)
  const float*          xrow  = x   + (size_t)row * ((size_t)S_ * I_);
  const unsigned short* xbrow = xbf ? xbf + (size_t)row * ((size_t)S_ * I_) : nullptr;

  const float bs_i = bsum[ 0 + lm];
  const float bs_f = bsum[16 + lm];
  const float bs_g = bsum[32 + lm];
  const float bs_o = bsum[48 + lm];

  v8f c = {};   // cell-state tile in C/D accumulator layout (M = r + 8*half + m0, N = n0 + lm)

  for (int t = 0; t < S_; ++t) {
    const unsigned short* hin  = (t & 1) ? h1buf : h0buf;
    unsigned short*       hout = (t & 1) ? h0buf : h1buf;

    v8f acc0 = {}, acc1 = {}, acc2 = {}, acc3 = {};

    // ---- Phase 1: recurrent  h_{t-1} @ W_hh^T   (K = 1024, bf16 h from L2)
    {
      const unsigned short* hrow = hin + row * H_;
      for (int kb = 0; kb < H_ / 32; ++kb) {
        Frag16 a;
        const int k0 = kb * 32 + (half ? 16 : 0);         // A layout: 16 contiguous K per lane
        a.q[0] = *(const uint4*)(hrow + k0);
        a.q[1] = *(const uint4*)(hrow + k0 + 8);
        const int kw = kb * 32 + (half ? 16 : 0);         // B layout: 16 contiguous K per lane
        Frag16 b0, b1, b2, b3;
        const unsigned short* w0 = Whh_s + ((0 * 16 + lm) << 10) + kw;
        const unsigned short* w1 = Whh_s + ((1 * 16 + lm) << 10) + kw;
        const unsigned short* w2 = Whh_s + ((2 * 16 + lm) << 10) + kw;
        const unsigned short* w3 = Whh_s + ((3 * 16 + lm) << 10) + kw;
        b0.q[0] = *(const uint4*)(w0); b0.q[1] = *(const uint4*)(w0 + 8);
        b1.q[0] = *(const uint4*)(w1); b1.q[1] = *(const uint4*)(w1 + 8);
        b2.q[0] = *(const uint4*)(w2); b2.q[1] = *(const uint4*)(w2 + 8);
        b3.q[0] = *(const uint4*)(w3); b3.q[1] = *(const uint4*)(w3 + 8);
        acc0 = __builtin_amdgcn_wmma_f32_16x16x32_bf16(false, a.v, false, b0.v, (short)0, acc0, false, false);
        acc1 = __builtin_amdgcn_wmma_f32_16x16x32_bf16(false, a.v, false, b1.v, (short)0, acc1, false, false);
        acc2 = __builtin_amdgcn_wmma_f32_16x16x32_bf16(false, a.v, false, b2.v, (short)0, acc2, false, false);
        acc3 = __builtin_amdgcn_wmma_f32_16x16x32_bf16(false, a.v, false, b3.v, (short)0, acc3, false, false);
      }
    }

    // ---- Phase 2: input  x_t @ W_ih^T   (K = 1024)
    if (xbrow) {
      // Fast path: x pre-converted to bf16 -> pure b128 loads, no cvt VALU in hot loop
      const unsigned short* xt = xbrow + (size_t)t * I_;
      for (int kb = 0; kb < I_ / 32; ++kb) {
        Frag16 a;
        const int k0 = kb * 32 + (half ? 16 : 0);
        a.q[0] = *(const uint4*)(xt + k0);
        a.q[1] = *(const uint4*)(xt + k0 + 8);
        const int kw = kb * 32 + (half ? 16 : 0);
        Frag16 b0, b1, b2, b3;
        const unsigned short* w0 = Wih_s + ((0 * 16 + lm) << 10) + kw;
        const unsigned short* w1 = Wih_s + ((1 * 16 + lm) << 10) + kw;
        const unsigned short* w2 = Wih_s + ((2 * 16 + lm) << 10) + kw;
        const unsigned short* w3 = Wih_s + ((3 * 16 + lm) << 10) + kw;
        b0.q[0] = *(const uint4*)(w0); b0.q[1] = *(const uint4*)(w0 + 8);
        b1.q[0] = *(const uint4*)(w1); b1.q[1] = *(const uint4*)(w1 + 8);
        b2.q[0] = *(const uint4*)(w2); b2.q[1] = *(const uint4*)(w2 + 8);
        b3.q[0] = *(const uint4*)(w3); b3.q[1] = *(const uint4*)(w3 + 8);
        acc0 = __builtin_amdgcn_wmma_f32_16x16x32_bf16(false, a.v, false, b0.v, (short)0, acc0, false, false);
        acc1 = __builtin_amdgcn_wmma_f32_16x16x32_bf16(false, a.v, false, b1.v, (short)0, acc1, false, false);
        acc2 = __builtin_amdgcn_wmma_f32_16x16x32_bf16(false, a.v, false, b2.v, (short)0, acc2, false, false);
        acc3 = __builtin_amdgcn_wmma_f32_16x16x32_bf16(false, a.v, false, b3.v, (short)0, acc3, false, false);
      }
      if (t + 1 < S_) {                   // warm L2/WGP$ for next timestep's x rows (bf16: 2KB/row)
        __builtin_prefetch(xt + I_, 0, 1);
        __builtin_prefetch(xt + I_ + 512, 0, 1);
      }
    } else {
      // Fallback: convert fp32 x in-flight (used only if d_ws is too small for x_bf16)
      const float* xt = xrow + (size_t)t * I_;
      for (int kb = 0; kb < I_ / 32; ++kb) {
        Frag16 a;
        const int k0 = kb * 32 + (half ? 16 : 0);
        a.q[0] = pack8bf(*(const float4*)(xt + k0),      *(const float4*)(xt + k0 + 4));
        a.q[1] = pack8bf(*(const float4*)(xt + k0 + 8),  *(const float4*)(xt + k0 + 12));
        const int kw = kb * 32 + (half ? 16 : 0);
        Frag16 b0, b1, b2, b3;
        const unsigned short* w0 = Wih_s + ((0 * 16 + lm) << 10) + kw;
        const unsigned short* w1 = Wih_s + ((1 * 16 + lm) << 10) + kw;
        const unsigned short* w2 = Wih_s + ((2 * 16 + lm) << 10) + kw;
        const unsigned short* w3 = Wih_s + ((3 * 16 + lm) << 10) + kw;
        b0.q[0] = *(const uint4*)(w0); b0.q[1] = *(const uint4*)(w0 + 8);
        b1.q[0] = *(const uint4*)(w1); b1.q[1] = *(const uint4*)(w1 + 8);
        b2.q[0] = *(const uint4*)(w2); b2.q[1] = *(const uint4*)(w2 + 8);
        b3.q[0] = *(const uint4*)(w3); b3.q[1] = *(const uint4*)(w3 + 8);
        acc0 = __builtin_amdgcn_wmma_f32_16x16x32_bf16(false, a.v, false, b0.v, (short)0, acc0, false, false);
        acc1 = __builtin_amdgcn_wmma_f32_16x16x32_bf16(false, a.v, false, b1.v, (short)0, acc1, false, false);
        acc2 = __builtin_amdgcn_wmma_f32_16x16x32_bf16(false, a.v, false, b2.v, (short)0, acc2, false, false);
        acc3 = __builtin_amdgcn_wmma_f32_16x16x32_bf16(false, a.v, false, b3.v, (short)0, acc3, false, false);
      }
      if (t + 1 < S_) {
        __builtin_prefetch(xt + I_, 0, 1);
        __builtin_prefetch(xt + I_ + 256, 0, 1);
        __builtin_prefetch(xt + I_ + 512, 0, 1);
        __builtin_prefetch(xt + I_ + 768, 0, 1);
      }
    }

    // ---- Elementwise LSTM cell update (torch gate order i,f,g,o); c stays in registers
    float hn[8];
#pragma unroll
    for (int r = 0; r < 8; ++r) {
      const float ig = sig_(acc0[r] + bs_i);
      const float fg = sig_(acc1[r] + bs_f);
      const float gg = tanh_(acc2[r] + bs_g);
      const float og = sig_(acc3[r] + bs_o);
      const float cn = fg * c[r] + ig * gg;
      c[r]  = cn;
      hn[r] = og * tanh_(cn);
    }

    // ---- Write h (bf16 ping-pong) and the fp32 output sequence
    const int mbase = m0 + (half ? 8 : 0);
#pragma unroll
    for (int r = 0; r < 8; ++r) {
      const int mrow = mbase + r;
      hout[mrow * H_ + n0 + lm] = f2bf(hn[r]);
      out[(size_t)mrow * ((size_t)S_ * H_) + (size_t)t * H_ + (n0 + lm)] = hn[r];
    }

    // ---- Device-wide barrier: all 64 persistent blocks step in lockstep
    __threadfence();
    __syncthreads();
    if (tid == 0) {
      const unsigned g = __hip_atomic_load(barGen, __ATOMIC_ACQUIRE, __HIP_MEMORY_SCOPE_AGENT);
      const unsigned arrived =
          __hip_atomic_fetch_add(barCnt, 1u, __ATOMIC_ACQ_REL, __HIP_MEMORY_SCOPE_AGENT);
      if (arrived == NBLK - 1) {
        __hip_atomic_store(barCnt, 0u, __ATOMIC_RELAXED, __HIP_MEMORY_SCOPE_AGENT);
        __hip_atomic_fetch_add(barGen, 1u, __ATOMIC_ACQ_REL, __HIP_MEMORY_SCOPE_AGENT);
      } else {
        while (__hip_atomic_load(barGen, __ATOMIC_ACQUIRE, __HIP_MEMORY_SCOPE_AGENT) == g) {
          __builtin_amdgcn_s_sleep(2);
        }
      }
    }
    __syncthreads();
  }
}

extern "C" void kernel_launch(void* const* d_in, const int* in_sizes, int n_in,
                              void* d_out, int out_size, void* d_ws, size_t ws_size,
                              hipStream_t stream) {
  (void)in_sizes; (void)n_in; (void)out_size;
  const float* x   = (const float*)d_in[0];
  const float* Wih = (const float*)d_in[1];
  const float* Whh = (const float*)d_in[2];
  const float* bih = (const float*)d_in[3];
  const float* bhh = (const float*)d_in[4];
  float* out = (float*)d_out;

  unsigned char* ws = (unsigned char*)d_ws;
  unsigned*       barCnt = (unsigned*)(ws + 0);
  unsigned*       barGen = (unsigned*)(ws + 64);
  unsigned short* h0     = (unsigned short*)(ws + 256);
  unsigned short* h1     = (unsigned short*)(ws + 256 + (size_t)B_ * H_ * 2);

  const size_t xbf_off   = 256 + 2 * (size_t)B_ * H_ * 2;        // after h ping-pong
  const size_t xbf_bytes = (size_t)B_ * S_ * I_ * 2;             // 128 MB
  const bool   have_xbf  = ws_size >= xbf_off + xbf_bytes;
  unsigned short* xbf    = have_xbf ? (unsigned short*)(ws + xbf_off) : nullptr;

  // Zero barrier state + h0 (c0 lives in registers; h1 is written before first read).
  hipMemsetAsync(d_ws, 0, 256 + (size_t)B_ * H_ * 2, stream);

  if (have_xbf) {
    // 64M elements, 8 per thread-iteration; purely bandwidth-bound one-shot pass.
    x_to_bf16<<<8192, 256, 0, stream>>>(x, xbf);
  }

  const size_t shmem = (size_t)64 * 1024 * 2 * 2 + 256;  // 256KB weights + bias sums
  hipFuncSetAttribute((const void*)lstm_persistent,
                      hipFuncAttributeMaxDynamicSharedMemorySize, (int)shmem);

  lstm_persistent<<<NBLK, TPB, shmem, stream>>>(x, Wih, Whh, bih, bhh, out,
                                                h0, h1, xbf, barCnt, barGen);
}
